// MultiHeadDifferentialAttention_76742475645023
// MI455X (gfx1250) — compile-verified, matrix operand order checked by reading, not run on hardware
//
#include <hip/hip_runtime.h>
#include <hip/hip_bf16.h>

typedef __attribute__((ext_vector_type(16))) __bf16 bf16x16;
typedef __attribute__((ext_vector_type(8)))  __bf16 bf16x8;
typedef __attribute__((ext_vector_type(8)))  float  f32x8;

#define D_MODEL 4096
#define TWO_D   8192
#define SEQ     512
#define BATCH   8
#define NROW    4096          /* B*S */
#define SCALE   0.04419417382415922f   /* 1/sqrt(512) */
#define LAMBDA_INIT 0.8f

__device__ __forceinline__ f32x8 wmma_bf16(bf16x16 a, bf16x16 b, f32x8 c) {
  return __builtin_amdgcn_wmma_f32_16x16x32_bf16(false, a, false, b, (short)0, c,
                                                 false, false);
}

// ---- CDNA5 async global->LDS copy (16 bytes per lane), tracked by ASYNCcnt.
__device__ __forceinline__ void async_g2l_b128(void* lds, const void* gaddr) {
  unsigned l = (unsigned)(uintptr_t)lds;   // low 32 bits = LDS offset
  asm volatile("global_load_async_to_lds_b128 %0, %1, off"
               :: "v"(l), "v"(gaddr) : "memory");
}
__device__ __forceinline__ void wait_async0() {
#if __has_builtin(__builtin_amdgcn_s_wait_asynccnt)
  __builtin_amdgcn_s_wait_asynccnt(0);
#else
  asm volatile("s_wait_asynccnt 0x0" ::: "memory");
#endif
}

// Load a 16x32 bf16 fragment from a row-major [16][ld] bf16 tile.
// Per ISA 7.12.2: lanes 0-15 -> K {0..7,16..23}, lanes 16-31 -> K {8..15,24..31}.
__device__ __forceinline__ bf16x16 load_frag(const __bf16* base, int ld) {
  const int lane = threadIdx.x & 31;
  const int m  = lane & 15;
  const int kb = (lane & 16) ? 8 : 0;
  const __bf16* p = base + (size_t)m * ld + kb;
  bf16x8 lo = *(const bf16x8*)(p);        // K = kb .. kb+7
  bf16x8 hi = *(const bf16x8*)(p + 16);   // K = kb+16 .. kb+23
  return __builtin_shufflevector(lo, hi, 0,1,2,3,4,5,6,7,8,9,10,11,12,13,14,15);
}

// ---------------------------------------------------------------------------
// Kernel 0: one-shot f32 -> bf16 conversion (bandwidth-bound).
// n must be a multiple of 2048 (true for all tensors here).
// ---------------------------------------------------------------------------
__global__ __launch_bounds__(256)
void cvt_f32_bf16_kernel(const float* __restrict__ src, __bf16* __restrict__ dst)
{
  const size_t i = (size_t)blockIdx.x * 256 + threadIdx.x;
  const float4* s = (const float4*)src + 2 * i;
  float4 a = s[0], b = s[1];
  bf16x8 o;
  o[0]=(__bf16)a.x; o[1]=(__bf16)a.y; o[2]=(__bf16)a.z; o[3]=(__bf16)a.w;
  o[4]=(__bf16)b.x; o[5]=(__bf16)b.y; o[6]=(__bf16)b.z; o[7]=(__bf16)b.w;
  *((bf16x8*)dst + i) = o;
}

// ---------------------------------------------------------------------------
// Kernel 1: q/k/v projections (bf16 in, bf16 out).
//   C[4096x8192] = Xbf[4096x4096] @ Wbf^T + bias
// Block tile 64x256, K-step 32, wave tile 32x64 (2x4 WMMA tiles),
// async bf16 tiles + double-buffered LDS.
// z==2 (the V projection) is stored transposed Vt[n][row] so kernel 3's
// B tiles become k-contiguous.
// ---------------------------------------------------------------------------
__global__ __launch_bounds__(256)
void proj_gemm_kernel(const __bf16* __restrict__ X,
                      const __bf16* __restrict__ Wq, const float* __restrict__ Bq,
                      const __bf16* __restrict__ Wk, const float* __restrict__ Bk,
                      const __bf16* __restrict__ Wv, const float* __restrict__ Bv,
                      __bf16* __restrict__ Oq, __bf16* __restrict__ Ok,
                      __bf16* __restrict__ OvT)
{
  const __bf16* W; const float* Bias; __bf16* O;
  if (blockIdx.z == 0)      { W = Wq; Bias = Bq; O = Oq;  }
  else if (blockIdx.z == 1) { W = Wk; Bias = Bk; O = Ok;  }
  else                      { W = Wv; Bias = Bv; O = OvT; }
  const bool transOut = (blockIdx.z == 2);

  __shared__ __bf16 As[2][64][40];    // [m][k], 80B rows (16B aligned)
  __shared__ __bf16 Bs[2][256][40];   // [n][k] (W rows are k-contiguous)

  const int tid = threadIdx.x;
  const int wid = tid >> 5;
  const int m0 = blockIdx.y * 64;
  const int n0 = blockIdx.x * 256;
  const int wm = (wid >> 2) * 32;   // 0 / 32
  const int wn = (wid & 3) * 64;    // 0..192

  f32x8 acc[2][4] = {};

  const int arow = tid >> 2, aseg = (tid & 3) * 8;   // 64 x 32, 8 bf16/thread
  const int brow = tid;                              // 256 x 32, 32 bf16/thread

  const __bf16* aG = X + (size_t)(m0 + arow) * D_MODEL + aseg;
  const __bf16* bG = W + (size_t)(n0 + brow) * D_MODEL;

  // prologue: stage k0 = 0 into buffer 0
  async_g2l_b128(&As[0][arow][aseg], aG);
  async_g2l_b128(&Bs[0][brow][0],  bG);
  async_g2l_b128(&Bs[0][brow][8],  bG + 8);
  async_g2l_b128(&Bs[0][brow][16], bG + 16);
  async_g2l_b128(&Bs[0][brow][24], bG + 24);

  int cur = 0;
  for (int k0 = 0; k0 < D_MODEL; k0 += 32, cur ^= 1) {
    wait_async0();
    __syncthreads();                       // buffer `cur` ready for all waves
    if (k0 + 32 < D_MODEL) {               // overlap next copy with compute
      const int nb = cur ^ 1;
      const __bf16* an = aG + k0 + 32;
      const __bf16* bn = bG + k0 + 32;
      async_g2l_b128(&As[nb][arow][aseg], an);
      async_g2l_b128(&Bs[nb][brow][0],  bn);
      async_g2l_b128(&Bs[nb][brow][8],  bn + 8);
      async_g2l_b128(&Bs[nb][brow][16], bn + 16);
      async_g2l_b128(&Bs[nb][brow][24], bn + 24);
    }
    bf16x16 af0 = load_frag(&As[cur][wm][0], 40);
    bf16x16 af1 = load_frag(&As[cur][wm + 16][0], 40);
    #pragma unroll
    for (int j = 0; j < 4; ++j) {
      bf16x16 bf = load_frag(&Bs[cur][wn + j*16][0], 40);
      acc[0][j] = wmma_bf16(af0, bf, acc[0][j]);
      acc[1][j] = wmma_bf16(af1, bf, acc[1][j]);
    }
  }

  const int lane = tid & 31;
  const int nl = lane & 15;
  const int mh = (lane & 16) ? 8 : 0;
  #pragma unroll
  for (int i = 0; i < 2; ++i) {
    #pragma unroll
    for (int j = 0; j < 4; ++j) {
      const int col = n0 + wn + j*16 + nl;
      const float bias = Bias[col];
      const int row = m0 + wm + i*16 + mh;
      if (transOut) {
        #pragma unroll
        for (int v = 0; v < 8; ++v)
          O[(size_t)col * NROW + row + v] = (__bf16)(acc[i][j][v] + bias);
      } else {
        #pragma unroll
        for (int v = 0; v < 8; ++v)
          O[(size_t)(row + v) * TWO_D + col] = (__bf16)(acc[i][j][v] + bias);
      }
    }
  }
}

// ---------------------------------------------------------------------------
// Kernel 2: per (batch, 16-query tile): S1=q1@k1^T, S2=q2@k2^T (scaled),
// row softmax of both, P = a1 - lam[key]*a2 -> bf16.
// ---------------------------------------------------------------------------
__global__ __launch_bounds__(256)
void diff_scores_kernel(const __bf16* __restrict__ Q, const __bf16* __restrict__ K,
                        const float* __restrict__ lq1, const float* __restrict__ lk1,
                        const float* __restrict__ lq2, const float* __restrict__ lk2,
                        __bf16* __restrict__ P)
{
  __shared__ float S1[16][SEQ];
  __shared__ float S2[16][SEQ];
  __shared__ float lam[SEQ];
  __shared__ float red[4][16][16];

  const int tid = threadIdx.x;
  const int wid = tid >> 5;
  const int b  = blockIdx.y;
  const int r0 = blockIdx.x * 16;
  const size_t qbase = ((size_t)b * SEQ + r0) * TWO_D;
  const size_t kbase = ((size_t)b * SEQ) * TWO_D;

  {
    const int i = tid;
    lam[i]       = __expf(lq1[i]*lk1[i])         - __expf(lq2[i]*lk2[i])         + LAMBDA_INIT;
    lam[i + 256] = __expf(lq1[i+256]*lk1[i+256]) - __expf(lq2[i+256]*lk2[i+256]) + LAMBDA_INIT;
  }

  f32x8 acc1[4] = {};
  f32x8 acc2[4] = {};
  for (int k0 = 0; k0 < D_MODEL; k0 += 32) {
    bf16x16 a1 = load_frag(Q + qbase + k0, TWO_D);            // q1 rows
    bf16x16 a2 = load_frag(Q + qbase + D_MODEL + k0, TWO_D);  // q2 rows
    #pragma unroll
    for (int t = 0; t < 4; ++t) {
      const size_t kb = kbase + (size_t)(wid*64 + t*16) * TWO_D;
      bf16x16 b1 = load_frag(K + kb + k0, TWO_D);             // k1 rows (A@B^T)
      acc1[t] = wmma_bf16(a1, b1, acc1[t]);
      bf16x16 b2 = load_frag(K + kb + D_MODEL + k0, TWO_D);   // k2 rows
      acc2[t] = wmma_bf16(a2, b2, acc2[t]);
    }
  }

  {
    const int lane = tid & 31;
    const int nl = lane & 15;
    const int mh = (lane & 16) ? 8 : 0;
    #pragma unroll
    for (int t = 0; t < 4; ++t) {
      const int c = wid*64 + t*16 + nl;
      #pragma unroll
      for (int v = 0; v < 8; ++v) {
        S1[mh + v][c] = acc1[t][v] * SCALE;
        S2[mh + v][c] = acc2[t][v] * SCALE;
      }
    }
  }
  __syncthreads();

  // cooperative softmax: 16 threads per row, 32 cols each
  const int r  = tid >> 4;
  const int sl = tid & 15;
  const int cs = sl * 32;

  float m1 = -1e30f, m2 = -1e30f;
  for (int j = 0; j < 32; ++j) {
    m1 = fmaxf(m1, S1[r][cs + j]);
    m2 = fmaxf(m2, S2[r][cs + j]);
  }
  red[0][r][sl] = m1; red[1][r][sl] = m2;
  __syncthreads();
  float rm1 = red[0][r][0], rm2 = red[1][r][0];
  for (int j = 1; j < 16; ++j) {
    rm1 = fmaxf(rm1, red[0][r][j]);
    rm2 = fmaxf(rm2, red[1][r][j]);
  }

  float s1 = 0.f, s2 = 0.f;
  for (int j = 0; j < 32; ++j) {
    float e1 = __expf(S1[r][cs + j] - rm1); S1[r][cs + j] = e1; s1 += e1;
    float e2 = __expf(S2[r][cs + j] - rm2); S2[r][cs + j] = e2; s2 += e2;
  }
  red[2][r][sl] = s1; red[3][r][sl] = s2;
  __syncthreads();
  float rs1 = 0.f, rs2 = 0.f;
  for (int j = 0; j < 16; ++j) { rs1 += red[2][r][j]; rs2 += red[3][r][j]; }
  const float i1 = 1.f / rs1, i2 = 1.f / rs2;

  __bf16* Prow = P + ((size_t)b * SEQ + (r0 + r)) * SEQ;
  for (int j = 0; j < 32; ++j) {
    const int c = cs + j;
    const float p = S1[r][c] * i1 - lam[c] * (S2[r][c] * i2);
    Prow[c] = (__bf16)p;
  }
}

// ---------------------------------------------------------------------------
// Kernel 3: out[b] = P[b][512x512] @ v[b][512x8192] -> f32
// B sourced from the transposed V (Vt[n][row]) so its tiles are k-contiguous.
// Block tile 64x128, K-step 32; async bf16 tiles, double-buffered LDS.
// ---------------------------------------------------------------------------
__global__ __launch_bounds__(256)
void out_gemm_kernel(const __bf16* __restrict__ P, const __bf16* __restrict__ Vt,
                     float* __restrict__ Out)
{
  __shared__ __bf16 As[2][64][40];    // [m][k]
  __shared__ __bf16 Bs[2][128][40];   // [n][k]

  const int tid = threadIdx.x;
  const int wid = tid >> 5;
  const int b  = blockIdx.z;
  const int m0 = blockIdx.y * 64;
  const int n0 = blockIdx.x * 128;
  const int wm = (wid >> 2) * 32;
  const int wn = (wid & 3) * 32;

  const __bf16* Pb = P + (size_t)b * SEQ * SEQ;

  f32x8 acc[2][2] = {};

  const int arow = tid >> 2, aseg = (tid & 3) * 8;   // 64 x 32 A tile
  const int brow = tid >> 1, bseg = (tid & 1) * 16;  // 128 x 32 B tile

  const __bf16* aG = Pb + (size_t)(m0 + arow) * SEQ + aseg;
  // Vt[n][row]: rows of length NROW; batch b occupies cols b*SEQ .. b*SEQ+511
  const __bf16* bG = Vt + (size_t)(n0 + brow) * NROW + (size_t)b * SEQ + bseg;

  async_g2l_b128(&As[0][arow][aseg],   aG);
  async_g2l_b128(&Bs[0][brow][bseg],   bG);
  async_g2l_b128(&Bs[0][brow][bseg+8], bG + 8);

  int cur = 0;
  for (int k0 = 0; k0 < SEQ; k0 += 32, cur ^= 1) {
    wait_async0();
    __syncthreads();
    if (k0 + 32 < SEQ) {
      const int nb = cur ^ 1;
      const __bf16* an = aG + k0 + 32;
      const __bf16* bn = bG + k0 + 32;
      async_g2l_b128(&As[nb][arow][aseg],   an);
      async_g2l_b128(&Bs[nb][brow][bseg],   bn);
      async_g2l_b128(&Bs[nb][brow][bseg+8], bn + 8);
    }
    #pragma unroll
    for (int i = 0; i < 2; ++i) {
      bf16x16 af = load_frag(&As[cur][wm + i*16][0], 40);
      #pragma unroll
      for (int j = 0; j < 2; ++j) {
        bf16x16 bf = load_frag(&Bs[cur][wn + j*16][0], 40);
        acc[i][j] = wmma_bf16(af, bf, acc[i][j]);
      }
    }
  }

  const int lane = tid & 31;
  const int nl = lane & 15;
  const int mh = (lane & 16) ? 8 : 0;
  #pragma unroll
  for (int i = 0; i < 2; ++i) {
    #pragma unroll
    for (int j = 0; j < 2; ++j) {
      const int col = n0 + wn + j*16 + nl;
      const int row = m0 + wm + i*16 + mh;
      #pragma unroll
      for (int v = 0; v < 8; ++v)
        Out[((size_t)b * SEQ + row + v) * TWO_D + col] = acc[i][j][v];
    }
  }
}

// ---------------------------------------------------------------------------
extern "C" void kernel_launch(void* const* d_in, const int* in_sizes, int n_in,
                              void* d_out, int out_size, void* d_ws, size_t ws_size,
                              hipStream_t stream) {
  const float* x   = (const float*)d_in[0];
  const float* wq  = (const float*)d_in[1];
  const float* bq  = (const float*)d_in[2];
  const float* wk  = (const float*)d_in[3];
  const float* bk  = (const float*)d_in[4];
  const float* wv  = (const float*)d_in[5];
  const float* bv  = (const float*)d_in[6];
  const float* lq1 = (const float*)d_in[7];
  const float* lk1 = (const float*)d_in[8];
  const float* lq2 = (const float*)d_in[9];
  const float* lk2 = (const float*)d_in[10];
  float* out = (float*)d_out;

  __bf16* ws = (__bf16*)d_ws;
  const size_t NX = (size_t)NROW * D_MODEL;    // 16,777,216
  const size_t NW = (size_t)TWO_D * D_MODEL;   // 33,554,432
  const size_t QKV = (size_t)NROW * TWO_D;     // 33,554,432

  __bf16* Xbf  = ws;
  __bf16* Wqbf = Xbf + NX;
  __bf16* Wkbf = Wqbf + NW;
  __bf16* Wvbf = Wkbf + NW;
  __bf16* Q    = Wvbf + NW;
  __bf16* K    = Q + QKV;
  __bf16* VT   = K + QKV;       // transposed V: [TWO_D][NROW]
  __bf16* P    = VT + QKV;      // B*S*S bf16

  // 0) one-shot f32 -> bf16 conversions (each count is a multiple of 2048)
  cvt_f32_bf16_kernel<<<dim3((unsigned)(NX / 2048)), 256, 0, stream>>>(x, Xbf);
  cvt_f32_bf16_kernel<<<dim3((unsigned)(NW / 2048)), 256, 0, stream>>>(wq, Wqbf);
  cvt_f32_bf16_kernel<<<dim3((unsigned)(NW / 2048)), 256, 0, stream>>>(wk, Wkbf);
  cvt_f32_bf16_kernel<<<dim3((unsigned)(NW / 2048)), 256, 0, stream>>>(wv, Wvbf);

  // 1) projections (z selects q/k/v weight set; V stored transposed)
  proj_gemm_kernel<<<dim3(TWO_D/256, NROW/64, 3), 256, 0, stream>>>(
      Xbf, Wqbf, bq, Wkbf, bk, Wvbf, bv, Q, K, VT);

  // 2) differential softmax scores
  diff_scores_kernel<<<dim3(SEQ/16, BATCH), 256, 0, stream>>>(
      Q, K, lq1, lk1, lq2, lk2, P);

  // 3) output GEMM
  out_gemm_kernel<<<dim3(TWO_D/128, SEQ/64, BATCH), 256, 0, stream>>>(
      P, VT, out);
}
